// LocalSelfAttentionFusion_87024627352253
// MI455X (gfx1250) — matrix operand order repaired and checked by reading, compiled, not run on hardware
//
#include <hip/hip_runtime.h>
#include <hip/hip_bf16.h>
#include <math.h>

constexpr int kD   = 120;   // feature dim
constexpr int kHD  = 15;    // head dim
constexpr int kMT  = 2;     // M-tiles (16 rows each) per wave
constexpr int kNW  = 2;     // waves per block
constexpr int kWQR = 368;   // 3*D=360 padded to 16-multiple
constexpr int kWK  = 128;   // K = 120 padded to 32-multiple
constexpr int kOWR = 128;   // out_w rows padded 120 -> 128

typedef __attribute__((ext_vector_type(16))) _Float16 v16h;
typedef __attribute__((ext_vector_type(8)))  _Float16 v8h;
typedef __attribute__((ext_vector_type(8)))  float    v8f;
typedef __attribute__((ext_vector_type(4)))  float    v4f;

// ---------------------------------------------------------------------------
// Pre-kernel: convert fp32 weights into zero-padded f16 tiles in workspace.
//   wqkv_h : [368][128] f16   (row n = output col n of qkv, cols = K)
//   outw_h : [128][128] f16
// ---------------------------------------------------------------------------
__global__ void convert_weights_k(const float* __restrict__ in_proj_w,
                                  const float* __restrict__ out_w,
                                  _Float16* __restrict__ wqkv_h,
                                  _Float16* __restrict__ outw_h)
{
    const int total1 = kWQR * kWK;
    const int total2 = kOWR * kWK;
    for (int i = blockIdx.x * blockDim.x + threadIdx.x; i < total1 + total2;
         i += gridDim.x * blockDim.x) {
        if (i < total1) {
            int n = i / kWK, k = i % kWK;
            float v = (n < 3 * kD && k < kD) ? in_proj_w[n * kD + k] : 0.f;
            wqkv_h[i] = (_Float16)v;
        } else {
            int j = i - total1;
            int n = j / kWK, k = j % kWK;
            float v = (n < kD && k < kD) ? out_w[n * kD + k] : 0.f;
            outw_h[j] = (_Float16)v;
        }
    }
}

// ---------------------------------------------------------------------------
// Fused: QKV wmma -> 2x2 attention -> out-proj wmma -> layernorm -> pool
// One wave handles kMT 16-token tiles (32 tokens = 16 sequences): each B
// fragment (L2-resident weights) feeds kMT WMMAs; B loads double-buffered
// one N-tile ahead. unroll-2 + launch_bounds(.,4) keep VGPRs <= 256 so the
// scheduler cannot hoist the whole weight stream into registers.
// ---------------------------------------------------------------------------
__global__ __launch_bounds__(kNW * 32, 4)
void fused_attn_k(const float* __restrict__ voxel,
                  const float* __restrict__ cnn,
                  const _Float16* __restrict__ wqkv_h,
                  const float* __restrict__ in_proj_b,
                  const _Float16* __restrict__ outw_h,
                  const float* __restrict__ out_b,
                  const float* __restrict__ gamma,
                  const float* __restrict__ beta,
                  float* __restrict__ out)
{
    __shared__ _Float16 s_qkv[kNW][kMT * 16][kWQR]; // 46 KB: qkv per wave
    __shared__ _Float16 s_o[kNW][kMT * 16][kWK];    // 16 KB: attn out staging

    const int w    = threadIdx.x >> 5;
    const int lane = threadIdx.x & 31;
    const int n16  = lane & 15;   // row (A) / col (B,D) within 16-tile
    const int hf   = lane >> 4;   // lane half

    // first token row handled by this wave (grid sized exactly, no tail)
    const int tokBase = (blockIdx.x * kNW + w) * (kMT * 16);

    // ---- stage 1: A fragments (16x32 f16, ISA layout) straight from HBM ----
    v16h afrag[kMT][4];
#pragma unroll
    for (int t = 0; t < kMT; ++t) {
        const int row = tokBase + t * 16 + n16;
        const int seq = row >> 1;
        const float* src = (row & 1) ? (cnn + (size_t)seq * kD)
                                     : (voxel + (size_t)seq * kD);
#pragma unroll
        for (int kc = 0; kc < 4; ++kc) {
            const int k0 = kc * 32 + 8 * hf;   // first 8 K of this lane
            const int k1 = k0 + 16;            // next 8 K
            v4f x0 = *(const v4f*)(src + k0);
            v4f x1 = *(const v4f*)(src + k0 + 4);
            v16h a;
#pragma unroll
            for (int e = 0; e < 4; ++e) { a[e] = (_Float16)x0[e]; a[4 + e] = (_Float16)x1[e]; }
            if (k1 + 7 < kD) {
                v4f y0 = *(const v4f*)(src + k1);
                v4f y1 = *(const v4f*)(src + k1 + 4);
#pragma unroll
                for (int e = 0; e < 4; ++e) { a[8 + e] = (_Float16)y0[e]; a[12 + e] = (_Float16)y1[e]; }
            } else {
#pragma unroll
                for (int e = 0; e < 8; ++e) a[8 + e] = (_Float16)0.f;   // K pad
            }
            afrag[t][kc] = a;
        }
    }

    // ---- stage 2: QKV = X * Wqkv^T + b ----
    auto loadB = [&](const _Float16* wmat, int j, v16h* dst) {
        const _Float16* p = wmat + (size_t)(j * 16 + n16) * kWK + 8 * hf;
#pragma unroll
        for (int kc = 0; kc < 4; ++kc) {
            v8h lo = *(const v8h*)(p + kc * 32);
            v8h hi = *(const v8h*)(p + kc * 32 + 16);
            v16h b;
#pragma unroll
            for (int e = 0; e < 8; ++e) { b[e] = lo[e]; b[8 + e] = hi[e]; }
            dst[kc] = b;
        }
    };

    {
        v16h bbuf0[4], bbuf1[4];
        loadB(wqkv_h, 0, bbuf0);
#pragma unroll 2
        for (int j = 0; j < 23; ++j) {
            v16h* bc = (j & 1) ? bbuf1 : bbuf0;
            v16h* bn = (j & 1) ? bbuf0 : bbuf1;
            if (j < 22) loadB(wqkv_h, j + 1, bn);   // prefetch next N-tile
            v8f acc[kMT];
#pragma unroll
            for (int t = 0; t < kMT; ++t)
                acc[t] = v8f{0.f, 0.f, 0.f, 0.f, 0.f, 0.f, 0.f, 0.f};
#pragma unroll
            for (int kc = 0; kc < 4; ++kc) {
#pragma unroll
                for (int t = 0; t < kMT; ++t)
                    acc[t] = __builtin_amdgcn_wmma_f32_16x16x32_f16(
                                 false, afrag[t][kc], false, bc[kc],
                                 (short)0, acc[t], false, false);
            }
            const int n = j * 16 + n16;             // qkv output column
            const float bias = (n < 3 * kD) ? in_proj_b[n] : 0.f;
#pragma unroll
            for (int t = 0; t < kMT; ++t)
#pragma unroll
                for (int i = 0; i < 8; ++i)
                    s_qkv[w][t * 16 + 8 * hf + i][n] = (_Float16)(acc[t][i] + bias);
        }
    }
    __syncthreads();

    // ---- stage 3: 2-token attention; 16 seq x 8 heads = 128 tasks, 4/lane --
    for (int i = lane; i < (kMT * 16) * 8; i += 32)      // zero K-pad of o tile
        s_o[w][i >> 3][kD + (i & 7)] = (_Float16)0.f;

    const float scale = 0.25819888974716113f;            // 1/sqrt(15)
#pragma unroll
    for (int r = 0; r < 4; ++r) {
        const int t    = lane + 32 * r;
        const int s    = t >> 3;            // local sequence 0..15
        const int head = t & 7;
        const int r0 = 2 * s, r1 = 2 * s + 1;
        const int cq = head * kHD, ck = kD + head * kHD, cv = 2 * kD + head * kHD;
        float s00 = 0.f, s01 = 0.f, s10 = 0.f, s11 = 0.f;
        for (int d = 0; d < kHD; ++d) {
            float q0 = (float)s_qkv[w][r0][cq + d], q1 = (float)s_qkv[w][r1][cq + d];
            float k0 = (float)s_qkv[w][r0][ck + d], k1 = (float)s_qkv[w][r1][ck + d];
            s00 += q0 * k0; s01 += q0 * k1; s10 += q1 * k0; s11 += q1 * k1;
        }
        s00 *= scale; s01 *= scale; s10 *= scale; s11 *= scale;
        const float m0 = fmaxf(s00, s01), m1 = fmaxf(s10, s11);
        const float e00 = __expf(s00 - m0), e01 = __expf(s01 - m0);
        const float e10 = __expf(s10 - m1), e11 = __expf(s11 - m1);
        const float i0 = 1.f / (e00 + e01), i1 = 1.f / (e10 + e11);
        const float w00 = e00 * i0, w01 = e01 * i0;
        const float w10 = e10 * i1, w11 = e11 * i1;
        for (int d = 0; d < kHD; ++d) {
            float v0 = (float)s_qkv[w][r0][cv + d], v1 = (float)s_qkv[w][r1][cv + d];
            s_o[w][r0][cq + d] = (_Float16)(w00 * v0 + w01 * v1);
            s_o[w][r1][cq + d] = (_Float16)(w10 * v0 + w11 * v1);
        }
    }
    __syncthreads();

    // ---- stage 4+5 per tile: out-proj wmma, layernorm, pool, store ----
#pragma unroll 1
    for (int t = 0; t < kMT; ++t) {
        // A fragments of o from LDS
        v16h ofrag[4];
#pragma unroll
        for (int kc = 0; kc < 4; ++kc) {
            const int k0 = kc * 32 + 8 * hf;
            v8h lo = *(const v8h*)(&s_o[w][t * 16 + n16][k0]);
            v8h hi = *(const v8h*)(&s_o[w][t * 16 + n16][k0 + 16]);
            v16h a;
#pragma unroll
            for (int e = 0; e < 8; ++e) { a[e] = lo[e]; a[8 + e] = hi[e]; }
            ofrag[kc] = a;
        }

        v8f y[8];
        {
            v16h bbuf0[4], bbuf1[4];
            loadB(outw_h, 0, bbuf0);
#pragma unroll 2
            for (int j = 0; j < 8; ++j) {
                v16h* bc = (j & 1) ? bbuf1 : bbuf0;
                v16h* bn = (j & 1) ? bbuf0 : bbuf1;
                if (j < 7) loadB(outw_h, j + 1, bn);
                v8f acc = {0.f, 0.f, 0.f, 0.f, 0.f, 0.f, 0.f, 0.f};
#pragma unroll
                for (int kc = 0; kc < 4; ++kc)
                    acc = __builtin_amdgcn_wmma_f32_16x16x32_f16(
                              false, ofrag[kc], false, bc[kc],
                              (short)0, acc, false, false);
                const int col = j * 16 + n16;
                const float bias = (col < kD) ? out_b[col] : 0.f;
#pragma unroll
                for (int i = 0; i < 8; ++i) acc[i] += bias;
                y[j] = acc;
            }
        }

        // layernorm per row (reduce across the 16-lane half via shfl)
#pragma unroll
        for (int i = 0; i < 8; ++i) {
            float sum = 0.f, sq = 0.f;
#pragma unroll
            for (int j = 0; j < 8; ++j) { float v = y[j][i]; sum += v; sq += v * v; }
#pragma unroll
            for (int m = 1; m < 16; m <<= 1) {
                sum += __shfl_xor(sum, m, 32);
                sq  += __shfl_xor(sq,  m, 32);
            }
            const float mean = sum * (1.f / kD);
            const float var  = sq * (1.f / kD) - mean * mean;  // pads exact 0
            const float rstd = rsqrtf(var + 1e-5f);
#pragma unroll
            for (int j = 0; j < 8; ++j) y[j][i] = (y[j][i] - mean) * rstd;
        }

        // gamma/beta, pool token pairs (both tokens live in the same lane)
        const int seqbase = (tokBase + t * 16) >> 1;
#pragma unroll
        for (int j = 0; j < 8; ++j) {
            const int col = j * 16 + n16;
            if (col < kD) {
                const float g = gamma[col], bb = beta[col];
#pragma unroll
                for (int p = 0; p < 4; ++p) {
                    const float z0 = y[j][2 * p]     * g + bb;
                    const float z1 = y[j][2 * p + 1] * g + bb;
                    const int seq = seqbase + 4 * hf + p;
                    out[(size_t)seq * kD + col] = 0.5f * (z0 + z1);
                }
            }
        }
    }
}

// ---------------------------------------------------------------------------
extern "C" void kernel_launch(void* const* d_in, const int* in_sizes, int n_in,
                              void* d_out, int out_size, void* d_ws, size_t ws_size,
                              hipStream_t stream)
{
    const float* voxel = (const float*)d_in[0];
    const float* cnn   = (const float*)d_in[1];
    const float* in_w  = (const float*)d_in[2];
    const float* in_b  = (const float*)d_in[3];
    const float* out_w = (const float*)d_in[4];
    const float* out_b = (const float*)d_in[5];
    const float* gamma = (const float*)d_in[6];
    const float* beta  = (const float*)d_in[7];
    float* out = (float*)d_out;

    _Float16* wqkv_h = (_Float16*)d_ws;
    _Float16* outw_h = (_Float16*)((char*)d_ws + (size_t)kWQR * kWK * sizeof(_Float16));

    const int BN     = in_sizes[0] / kD;          // 131072
    const int tokens = 2 * BN;                    // 262144
    const int perBlk = kNW * kMT * 16;            // 64 tokens / block
    const int blocks = tokens / perBlk;           // 4096 full blocks, no tail

    convert_weights_k<<<64, 256, 0, stream>>>(in_w, out_w, wqkv_h, outw_h);
    fused_attn_k<<<blocks, kNW * 32, 0, stream>>>(voxel, cnn, wqkv_h, in_b,
                                                  outw_h, out_b, gamma, beta, out);
}
